// BnBinActiveConv2d_48369921688118
// MI455X (gfx1250) — 8-bit matrix operand order repaired and checked by reading, compiled, not run on hardware
//
#include <hip/hip_runtime.h>
#include <stdint.h>

typedef int v8i  __attribute__((ext_vector_type(8)));
typedef int v2i_ __attribute__((ext_vector_type(2)));

#define N_IMG 32
#define C_IN  256
#define C_OUT 256
#define H_IN  56
#define W_IN  56
#define H_OUT 54
#define W_OUT 54
#define HW_IN (H_IN*W_IN)
#define KSEG  36
#define XS_PAD_PIX 64

__global__ void bn_stats_kernel(const float* __restrict__ x,
                                const float* __restrict__ gamma,
                                const float* __restrict__ beta,
                                float* __restrict__ scale,
                                float* __restrict__ bias) {
    __shared__ float ssum[256];
    __shared__ float ssq[256];
    const int c = blockIdx.x;
    const int t = threadIdx.x;
    float s = 0.f, q = 0.f;
    for (int i = t; i < N_IMG * HW_IN; i += 256) {
        int n  = i / HW_IN;
        int hw = i - n * HW_IN;
        float v = x[(size_t)(n * C_IN + c) * HW_IN + hw];
        s += v; q += v * v;
    }
    ssum[t] = s; ssq[t] = q;
    __syncthreads();
    for (int off = 128; off > 0; off >>= 1) {
        if (t < off) { ssum[t] += ssum[t + off]; ssq[t] += ssq[t + off]; }
        __syncthreads();
    }
    if (t == 0) {
        const float inv = 1.0f / (float)(N_IMG * HW_IN);
        float mu  = ssum[0] * inv;
        float var = ssq[0] * inv - mu * mu;
        float sc  = gamma[c] * rsqrtf(var + 1e-4f);
        scale[c] = sc;
        bias[c]  = beta[c] - mu * sc;
    }
}

__global__ void alpha_kernel(const float* __restrict__ W,
                             float* __restrict__ alpha) {
    __shared__ float s[64];
    const int co = blockIdx.x;
    const int t  = threadIdx.x;
    float a = 0.f;
    for (int i = t; i < C_IN * 9; i += 64)
        a += fabsf(W[(size_t)co * C_IN * 9 + i]);
    s[t] = a;
    __syncthreads();
    for (int off = 32; off > 0; off >>= 1) {
        if (t < off) s[t] += s[t + off];
        __syncthreads();
    }
    if (t == 0) alpha[co] = s[0] * (1.0f / (C_IN * 9));
}

__global__ void pack_w_kernel(const float* __restrict__ W,
                              int8_t* __restrict__ Bp) {
    const int blk  = blockIdx.x;
    const int s    = blk >> 4;
    const int lane = threadIdx.x;
    const int co   = (blk & 15) * 16 + (lane & 15);
    int words[8];
#pragma unroll
    for (int v = 0; v < 8; ++v) {
        unsigned wd = 0;
#pragma unroll
        for (int b = 0; b < 4; ++b) {
            int K    = s * 64 + 16 * (v >> 1) + 8 * (lane >> 4) + 4 * (v & 1) + b;
            int kidx = K >> 8;
            int ci   = K & 255;
            int kh   = kidx / 3, kw = kidx - 3 * kh;
            float w  = W[(((size_t)co * C_IN + ci) * 3 + kh) * 3 + kw];
            unsigned byte = (w >= 0.f) ? 0x01u : 0xFFu;
            wd |= byte << (8 * b);
        }
        words[v] = (int)wd;
    }
    v8i val;
#pragma unroll
    for (int i = 0; i < 8; ++i) val[i] = words[i];
    *(v8i*)(Bp + ((size_t)blk * 32 + lane) * 32) = val;
}

#define SB_PITCH 272
__global__ void binarize_kernel(const float* __restrict__ x,
                                const float* __restrict__ scale,
                                const float* __restrict__ bias,
                                int8_t* __restrict__ xs,
                                float* __restrict__ A) {
    __shared__ int8_t sb[64 * SB_PITCH];
    __shared__ float  sacc[256];
    const int t  = threadIdx.x;
    const int px = t & 63;
    const int cg = t >> 6;
    const long P  = (long)blockIdx.x * 64 + px;
    const int n  = (int)(P / HW_IN);
    const int hw = (int)(P - (long)n * HW_IN);
    const float* xp = x + (size_t)n * C_IN * HW_IN + hw;
    float acc = 0.f;
#pragma unroll 4
    for (int j = 0; j < 64; ++j) {
        int c = cg * 64 + j;
        float v = fmaf(scale[c], xp[(size_t)c * HW_IN], bias[c]);
        sb[px * SB_PITCH + c] = (v >= 0.f) ? (int8_t)1 : (int8_t)-1;
        acc += fabsf(v);
    }
    sacc[t] = acc;
    __syncthreads();
    if (t < 64)
        A[P] = (sacc[t] + sacc[64 + t] + sacc[128 + t] + sacc[192 + t]) *
               (1.0f / 256.0f);
    const int px2 = t >> 2;
    const int ch  = (t & 3) * 64;
    const int4* src = (const int4*)(sb + px2 * SB_PITCH + ch);
    int4* dst = (int4*)(xs + ((size_t)blockIdx.x * 64 + px2) * 256 + ch);
    dst[0] = src[0]; dst[1] = src[1]; dst[2] = src[2]; dst[3] = src[3];
}

__global__ void kmap_kernel(const float* __restrict__ A,
                            float* __restrict__ K) {
    int idx = blockIdx.x * 256 + threadIdx.x;
    if (idx >= N_IMG * H_OUT * W_OUT) return;
    int ow  = idx % W_OUT;
    int tmp = idx / W_OUT;
    int oh  = tmp % H_OUT;
    int n   = tmp / H_OUT;
    const float* Ap = A + ((size_t)n * H_IN + oh) * W_IN + ow;
    float s = 0.f;
#pragma unroll
    for (int kh = 0; kh < 3; ++kh)
#pragma unroll
        for (int kw = 0; kw < 3; ++kw)
            s += Ap[kh * W_IN + kw];
    K[idx] = s * (1.0f / 9.0f);
}

__device__ __forceinline__ v8i load_a_frag(const int8_t* p) {
    v2i_ c0 = *(const v2i_*)(p);
    v2i_ c1 = *(const v2i_*)(p + 16);
    v2i_ c2 = *(const v2i_*)(p + 32);
    v2i_ c3 = *(const v2i_*)(p + 48);
    v8i a;
    a[0] = c0[0]; a[1] = c0[1];
    a[2] = c1[0]; a[3] = c1[1];
    a[4] = c2[0]; a[5] = c2[1];
    a[6] = c3[0]; a[7] = c3[1];
    return a;
}

__device__ __forceinline__ int a_step_off(int s) {
    int kidx = s >> 2;
    int kh   = kidx / 3;
    int kw   = kidx - 3 * kh;
    return (kh * W_IN + kw) * 256 + (s & 3) * 64;
}

__global__ void __launch_bounds__(512, 1)
bingemm_kernel(const int8_t* __restrict__ xs,
               const int8_t* __restrict__ Bp,
               const float* __restrict__ Kmap,
               const float* __restrict__ alpha,
               float* __restrict__ out) {
    const int oh   = blockIdx.x;
    const int n    = blockIdx.y;
    const int wave = threadIdx.x >> 5;
    const int lane = threadIdx.x & 31;
    const int pxg  = wave & 3;
    const int cog  = wave >> 2;
    const int m    = lane & 15;
    const int half = lane >> 4;
    const int ow_m = pxg * 16 + m;

    const int8_t* abase = xs +
        (((size_t)n * H_IN + oh) * W_IN + ow_m) * 256 + half * 8;
    const v8i* bbase = (const v8i*)Bp + ((size_t)cog * 4 * 32 + lane);

    v8i acc0 = {}, acc1 = {}, acc2 = {}, acc3 = {};

    v8i a0  = load_a_frag(abase);
    v8i b00 = bbase[0];
    v8i b01 = bbase[32];
    v8i b02 = bbase[64];
    v8i b03 = bbase[96];

#pragma unroll 1
    for (int s = 0; s < KSEG; s += 2) {
        const int s1 = s + 1;
        v8i a1 = load_a_frag(abase + a_step_off(s1));
        const v8i* bp1 = bbase + (size_t)s1 * 16 * 32;
        v8i b10 = bp1[0];
        v8i b11 = bp1[32];
        v8i b12 = bp1[64];
        v8i b13 = bp1[96];
        if (s + 2 < KSEG)
            __builtin_prefetch(bbase + (size_t)(s + 2) * 16 * 32, 0, 1);

        acc0 = __builtin_amdgcn_wmma_i32_16x16x64_iu8(true, a0, true, b00, acc0,
                                                      false, false);
        acc1 = __builtin_amdgcn_wmma_i32_16x16x64_iu8(true, a0, true, b01, acc1,
                                                      false, false);
        acc2 = __builtin_amdgcn_wmma_i32_16x16x64_iu8(true, a0, true, b02, acc2,
                                                      false, false);
        acc3 = __builtin_amdgcn_wmma_i32_16x16x64_iu8(true, a0, true, b03, acc3,
                                                      false, false);

        const int s2 = (s + 2 < KSEG) ? s + 2 : s;
        a0 = load_a_frag(abase + a_step_off(s2));
        const v8i* bp2 = bbase + (size_t)s2 * 16 * 32;
        b00 = bp2[0];
        b01 = bp2[32];
        b02 = bp2[64];
        b03 = bp2[96];
        if (s + 3 < KSEG)
            __builtin_prefetch(bbase + (size_t)(s + 3) * 16 * 32, 0, 1);

        acc0 = __builtin_amdgcn_wmma_i32_16x16x64_iu8(true, a1, true, b10, acc0,
                                                      false, false);
        acc1 = __builtin_amdgcn_wmma_i32_16x16x64_iu8(true, a1, true, b11, acc1,
                                                      false, false);
        acc2 = __builtin_amdgcn_wmma_i32_16x16x64_iu8(true, a1, true, b12, acc2,
                                                      false, false);
        acc3 = __builtin_amdgcn_wmma_i32_16x16x64_iu8(true, a1, true, b13, acc3,
                                                      false, false);
    }

    const float* kp = Kmap + ((size_t)n * H_OUT + oh) * W_OUT;
    v8i accs[4] = {acc0, acc1, acc2, acc3};
#pragma unroll
    for (int f = 0; f < 4; ++f) {
        const int co   = cog * 64 + f * 16 + m;
        const float av = alpha[co];
        float* op = out + (((size_t)n * C_OUT + co) * H_OUT + oh) * W_OUT;
#pragma unroll
        for (int r = 0; r < 8; ++r) {
            int M  = half ? (r + 8) : r;
            int ow = pxg * 16 + M;
            if (ow < W_OUT) {
                float v = (float)accs[f][r] * kp[ow] * av;
                op[ow] = v > 0.f ? v : 0.f;
            }
        }
    }
}

extern "C" void kernel_launch(void* const* d_in, const int* in_sizes, int n_in,
                              void* d_out, int out_size, void* d_ws, size_t ws_size,
                              hipStream_t stream) {
    const float* x     = (const float*)d_in[0];
    const float* gamma = (const float*)d_in[1];
    const float* beta  = (const float*)d_in[2];
    const float* W     = (const float*)d_in[3];
    float* out = (float*)d_out;

    char* ws = (char*)d_ws;
    size_t off = 0;
    auto alloc = [&](size_t bytes) {
        size_t o = off;
        off = (off + bytes + 255) & ~(size_t)255;
        return o;
    };
    float*  scale = (float*)(ws + alloc(C_IN * 4));
    float*  bias  = (float*)(ws + alloc(C_IN * 4));
    float*  alpha = (float*)(ws + alloc(C_OUT * 4));
    float*  Amap  = (float*)(ws + alloc((size_t)N_IMG * HW_IN * 4));
    float*  Kmap  = (float*)(ws + alloc((size_t)N_IMG * H_OUT * W_OUT * 4));
    int8_t* Bpack = (int8_t*)(ws + alloc((size_t)KSEG * 16 * 32 * 32));
    int8_t* xsbuf = (int8_t*)(ws + alloc((size_t)(N_IMG * HW_IN + XS_PAD_PIX) * 256));
    (void)ws_size; (void)in_sizes; (void)n_in; (void)out_size;

    bn_stats_kernel<<<C_IN, 256, 0, stream>>>(x, gamma, beta, scale, bias);
    alpha_kernel<<<C_OUT, 64, 0, stream>>>(W, alpha);
    pack_w_kernel<<<KSEG * 16, 32, 0, stream>>>(W, Bpack);
    binarize_kernel<<<(N_IMG * HW_IN) / 64, 256, 0, stream>>>(x, scale, bias,
                                                              xsbuf, Amap);
    kmap_kernel<<<(N_IMG * H_OUT * W_OUT + 255) / 256, 256, 0, stream>>>(Amap,
                                                                         Kmap);
    dim3 g(H_OUT, N_IMG);
    bingemm_kernel<<<g, 512, 0, stream>>>(xsbuf, Bpack, Kmap, alpha, out);
}
